// CrossAttentionModule_17643725652109
// MI455X (gfx1250) — matrix operand order repaired and checked by reading, not compiled, not run on hardware
//
#include <hip/hip_runtime.h>
#include <hip/hip_bf16.h>

#define HS 1024
#define NH 16
#define DH 64

typedef __attribute__((ext_vector_type(16))) __bf16 v16bf;
typedef __attribute__((ext_vector_type(8)))  __bf16 v8bf;
typedef __attribute__((ext_vector_type(8)))  float  v8f;

#if defined(__has_builtin)
#if __has_builtin(__builtin_amdgcn_global_load_async_to_lds_b128) && \
    __has_builtin(__builtin_amdgcn_s_wait_asynccnt)
#define USE_ASYNC_LDS 1
#endif
#endif

#ifdef USE_ASYNC_LDS
// Probe feedback: param type is a generic pointer to a 16-byte int vector.
typedef __attribute__((__vector_size__(4 * sizeof(int)))) int v4i_t;
#define GPTR(p) ((v4i_t*)(p))
#define LPTR(p) ((v4i_t*)(p))
#endif

static __device__ __forceinline__ v8f zero8() {
  v8f z;
#pragma unroll
  for (int i = 0; i < 8; ++i) z[i] = 0.0f;
  return z;
}

static __device__ __forceinline__ v16bf cat8(v8bf lo, v8bf hh) {
  v16bf r;
#pragma unroll
  for (int i = 0; i < 8; ++i) { r[i] = lo[i]; r[i + 8] = hh[i]; }
  return r;
}

// ---------------------------------------------------------------- fp32->bf16
__global__ void cvt_f32_bf16(const float* __restrict__ in,
                             __bf16* __restrict__ out, int n) {
  int i = (blockIdx.x * blockDim.x + threadIdx.x) * 4;
  if (i + 3 < n) {
    const float4 v = *(const float4*)(in + i);
    out[i + 0] = (__bf16)v.x;
    out[i + 1] = (__bf16)v.y;
    out[i + 2] = (__bf16)v.z;
    out[i + 3] = (__bf16)v.w;
  }
}

// ------------------------------------------------- C[M,HS] = A[M,HS]*W^T + b
// W stored [out_feature][in_feature]; tile BM=64 BN=128 BK=32; 8 waves/block,
// each wave computes a 32x32 patch (2x2 WMMA tiles of 16x16, K-step 32).
__global__ __launch_bounds__(256) void gemm_xwt_bf16(
    const __bf16* __restrict__ A, const __bf16* __restrict__ W,
    const float* __restrict__ bias, __bf16* __restrict__ C, int M) {
  __shared__ __bf16 As[64 * 32];
  __shared__ __bf16 Ws[128 * 32];
  const int tid  = threadIdx.x;
  const int lane = tid & 31;
  const int w    = tid >> 5;
  const int r    = lane & 15;
  const int hi   = lane >> 4;
  const int wm   = w >> 2, wn = w & 3;
  const int m0   = blockIdx.x * 64;
  const int n0   = blockIdx.y * 128;

  v8f acc[2][2];
#pragma unroll
  for (int i = 0; i < 2; ++i)
#pragma unroll
    for (int j = 0; j < 2; ++j) acc[i][j] = zero8();

  const int arow = tid >> 2, ach = tid & 3;   // A tile: 64 rows x (4 x 8 bf16)
  const int wrow = tid >> 1, wch = tid & 1;   // W tile: 128 rows x (2 x 16 bf16)

  for (int k0 = 0; k0 < HS; k0 += 32) {
    // prefetch next k-tile into cache while this one is consumed
    if (k0 + 32 < HS) {
      __builtin_prefetch(A + (size_t)(m0 + arow) * HS + (k0 + 32) + ach * 8, 0, 0);
      __builtin_prefetch(W + (size_t)(n0 + wrow) * HS + (k0 + 32) + wch * 16, 0, 0);
    }
#ifdef USE_ASYNC_LDS
    __syncthreads();  // prior-iteration LDS reads complete
    __builtin_amdgcn_global_load_async_to_lds_b128(
        GPTR(A + (size_t)(m0 + arow) * HS + k0 + ach * 8),
        LPTR(As + arow * 32 + ach * 8), 0, 0);
    __builtin_amdgcn_global_load_async_to_lds_b128(
        GPTR(W + (size_t)(n0 + wrow) * HS + k0 + wch * 16),
        LPTR(Ws + wrow * 32 + wch * 16), 0, 0);
    __builtin_amdgcn_global_load_async_to_lds_b128(
        GPTR(W + (size_t)(n0 + wrow) * HS + k0 + wch * 16 + 8),
        LPTR(Ws + wrow * 32 + wch * 16 + 8), 0, 0);
    __builtin_amdgcn_s_wait_asynccnt(0);
    __syncthreads();  // all waves' async tile writes visible
#else
    v8bf  av = *(const v8bf*)(A + (size_t)(m0 + arow) * HS + k0 + ach * 8);
    v16bf wv = *(const v16bf*)(W + (size_t)(n0 + wrow) * HS + k0 + wch * 16);
    __syncthreads();
    *(v8bf*)(As + arow * 32 + ach * 8)   = av;
    *(v16bf*)(Ws + wrow * 32 + wch * 16) = wv;
    __syncthreads();
#endif

    v16bf af[2], bfm[2];
#pragma unroll
    for (int ms = 0; ms < 2; ++ms) {
      int row = wm * 32 + ms * 16 + r;
      // 16-bit A layout: lanes<16 hold K 0..15, lanes>=16 hold K 16..31 (contiguous)
      af[ms] = *(const v16bf*)(As + row * 32 + hi * 16);
    }
#pragma unroll
    for (int ns = 0; ns < 2; ++ns) {
      int col = wn * 32 + ns * 16 + r;
      // B layout: lanes<16 col=r K0..15, lanes>=16 col=r K16..31 (contiguous)
      bfm[ns] = *(const v16bf*)(Ws + col * 32 + hi * 16);
    }
#pragma unroll
    for (int ms = 0; ms < 2; ++ms)
#pragma unroll
      for (int ns = 0; ns < 2; ++ns)
        acc[ms][ns] = __builtin_amdgcn_wmma_f32_16x16x32_bf16(
            false, af[ms], false, bfm[ns], (short)0, acc[ms][ns], false, false);
  }

#pragma unroll
  for (int ms = 0; ms < 2; ++ms)
#pragma unroll
    for (int ns = 0; ns < 2; ++ns) {
      int col = n0 + wn * 32 + ns * 16 + r;
      float bcol = bias[col];
#pragma unroll
      for (int j = 0; j < 8; ++j) {
        int row = m0 + wm * 32 + ms * 16 + j + 8 * hi;  // C layout
        C[(size_t)row * HS + col] = (__bf16)(acc[ms][ns][j] + bcol);
      }
    }
}

// ------------------------------------------------------------- flash attention
// One wave per 16-row Q tile; 32-key chunks; online softmax; fp32 accumulators.
__global__ __launch_bounds__(128) void flash_attn_bf16(
    const __bf16* __restrict__ Q, const __bf16* __restrict__ Kp,
    const __bf16* __restrict__ Vp, const int* __restrict__ mask,
    float* __restrict__ Out, int Sq, int Sk) {
  __shared__ __bf16 Pl[4 * 16 * 32];
  __shared__ __bf16 Vt[4 * 64 * 32];
  const int lane = threadIdx.x & 31;
  const int w    = threadIdx.x >> 5;
  const int r    = lane & 15;
  const int hi   = lane >> 4;
  const int h    = blockIdx.y;
  const int b    = blockIdx.z;
  const int q0   = blockIdx.x * 64 + w * 16;
  const size_t qrow0 = (size_t)b * Sq + q0;
  const size_t krow0 = (size_t)b * Sk;
  __bf16* Pw = Pl + w * (16 * 32);
  __bf16* Vw = Vt + w * (64 * 32);

  v16bf qf[2];
#pragma unroll
  for (int f = 0; f < 2; ++f) {
    const __bf16* qb = Q + (qrow0 + r) * HS + h * DH + f * 32;
    qf[f] = *(const v16bf*)(qb + hi * 16);
  }

  float mrow[8], lrow[8];
  v8f o[4];
#pragma unroll
  for (int j = 0; j < 8; ++j) { mrow[j] = -3.0e38f; lrow[j] = 0.0f; }
#pragma unroll
  for (int d = 0; d < 4; ++d) o[d] = zero8();

  const float scale = 0.125f;  // 1/sqrt(DH)

  for (int key0 = 0; key0 < Sk; key0 += 32) {
    // prefetch next chunk's K/V rows
    if (key0 + 32 < Sk) {
      __builtin_prefetch(Kp + (krow0 + key0 + 32 + lane) * HS + h * DH, 0, 0);
      __builtin_prefetch(Vp + (krow0 + key0 + 32 + lane) * HS + h * DH, 0, 0);
    }
    // stage V^T chunk in per-wave LDS: Vw[d][k], d=0..63, k=0..31
    {
      const __bf16* vr = Vp + (krow0 + key0 + lane) * HS + h * DH;
#pragma unroll
      for (int c4 = 0; c4 < 4; ++c4) {
        v16bf vv = *(const v16bf*)(vr + c4 * 16);
#pragma unroll
        for (int e = 0; e < 16; ++e) Vw[(c4 * 16 + e) * 32 + lane] = vv[e];
      }
    }
    // scores S[16 x 32] = Q * K^T  (two 16x16 n-tiles, 2 k-steps each)
    v8f s[2];
#pragma unroll
    for (int t = 0; t < 2; ++t) {
      const __bf16* kb = Kp + (krow0 + key0 + t * 16 + r) * HS + h * DH;
      v16bf k0v = *(const v16bf*)(kb + hi * 16);
      v16bf k1v = *(const v16bf*)(kb + 32 + hi * 16);
      v8f c = zero8();
      c = __builtin_amdgcn_wmma_f32_16x16x32_bf16(false, qf[0], false, k0v,
                                                  (short)0, c, false, false);
      c = __builtin_amdgcn_wmma_f32_16x16x32_bf16(false, qf[1], false, k1v,
                                                  (short)0, c, false, false);
      s[t] = c;
    }
    // scale + key mask (each lane owns one key column per tile)
    int mv0 = mask[krow0 + key0 + r];
    int mv1 = mask[krow0 + key0 + 16 + r];
#pragma unroll
    for (int j = 0; j < 8; ++j) {
      s[0][j] = (mv0 == 0) ? -1.0e9f : s[0][j] * scale;
      s[1][j] = (mv1 == 0) ? -1.0e9f : s[1][j] * scale;
    }
    // online softmax; xor masks 1..8 reduce within each 16-lane half,
    // matching the C-layout row split (rows j / j+8).
#pragma unroll
    for (int j = 0; j < 8; ++j) {
      float cm = fmaxf(s[0][j], s[1][j]);
#pragma unroll
      for (int ml = 8; ml >= 1; ml >>= 1) cm = fmaxf(cm, __shfl_xor(cm, ml, 32));
      float mn = fmaxf(mrow[j], cm);
      float alpha = __expf(mrow[j] - mn);
      mrow[j] = mn;
      float p0 = __expf(s[0][j] - mn);
      float p1 = __expf(s[1][j] - mn);
      s[0][j] = p0; s[1][j] = p1;
      float rs = p0 + p1;
#pragma unroll
      for (int ml = 8; ml >= 1; ml >>= 1) rs += __shfl_xor(rs, ml, 32);
      lrow[j] = lrow[j] * alpha + rs;
#pragma unroll
      for (int d = 0; d < 4; ++d) o[d][j] *= alpha;
    }
    // P -> LDS (bf16) then reload as A-fragment (per-wave LDS, DS in-order)
#pragma unroll
    for (int t = 0; t < 2; ++t)
#pragma unroll
      for (int j = 0; j < 8; ++j)
        Pw[(j + 8 * hi) * 32 + t * 16 + r] = (__bf16)s[t][j];
    v16bf pf;
    {
      pf = *(const v16bf*)(Pw + r * 32 + hi * 16);
    }
    // O += P * V  (4 d-tiles of 16)
#pragma unroll
    for (int d = 0; d < 4; ++d) {
      v16bf vf = *(const v16bf*)(Vw + (d * 16 + r) * 32 + hi * 16);
      o[d] = __builtin_amdgcn_wmma_f32_16x16x32_bf16(false, pf, false, vf,
                                                     (short)0, o[d], false, false);
    }
  }
#pragma unroll
  for (int j = 0; j < 8; ++j) {
    float inv = 1.0f / lrow[j];
#pragma unroll
    for (int d = 0; d < 4; ++d)
      Out[(qrow0 + j + 8 * hi) * HS + h * DH + d * 16 + r] = o[d][j] * inv;
  }
}

// --------------------------------------------------- residual + LayerNorm
__global__ __launch_bounds__(256) void add_ln(
    const float* __restrict__ x, const float* __restrict__ att,
    const float* __restrict__ g, const float* __restrict__ beta,
    float* __restrict__ out) {
  __shared__ float red[256];
  const int tid = threadIdx.x;
  const size_t base = (size_t)blockIdx.x * HS;
  float v[4];
  float s = 0.f;
#pragma unroll
  for (int i = 0; i < 4; ++i) {
    int c = tid * 4 + i;
    v[i] = x[base + c] + att[base + c];
    s += v[i];
  }
  red[tid] = s; __syncthreads();
  for (int off = 128; off > 0; off >>= 1) {
    if (tid < off) red[tid] += red[tid + off];
    __syncthreads();
  }
  float mu = red[0] * (1.0f / HS);
  __syncthreads();
  float sq = 0.f;
#pragma unroll
  for (int i = 0; i < 4; ++i) { float d = v[i] - mu; sq += d * d; }
  red[tid] = sq; __syncthreads();
  for (int off = 128; off > 0; off >>= 1) {
    if (tid < off) red[tid] += red[tid + off];
    __syncthreads();
  }
  float rstd = rsqrtf(red[0] * (1.0f / HS) + 1e-5f);
#pragma unroll
  for (int i = 0; i < 4; ++i) {
    int c = tid * 4 + i;
    out[base + c] = (v[i] - mu) * rstd * g[c] + beta[c];
  }
}

extern "C" void kernel_launch(void* const* d_in, const int* in_sizes, int n_in,
                              void* d_out, int out_size, void* d_ws, size_t ws_size,
                              hipStream_t stream) {
  (void)in_sizes; (void)n_in; (void)out_size; (void)ws_size;
  const float* cl    = (const float*)d_in[0];
  const float* ev    = (const float*)d_in[1];
  const int*   cmask = (const int*)d_in[2];
  const int*   emask = (const int*)d_in[3];
  const float* W_cq = (const float*)d_in[4];  const float* b_cq = (const float*)d_in[5];
  const float* W_ek = (const float*)d_in[6];  const float* b_ek = (const float*)d_in[7];
  const float* W_ev = (const float*)d_in[8];  const float* b_ev = (const float*)d_in[9];
  const float* W_eq = (const float*)d_in[10]; const float* b_eq = (const float*)d_in[11];
  const float* W_ck = (const float*)d_in[12]; const float* b_ck = (const float*)d_in[13];
  const float* W_cv = (const float*)d_in[14]; const float* b_cv = (const float*)d_in[15];
  const float* ln_g = (const float*)d_in[16]; const float* ln_b = (const float*)d_in[17];
  float* out = (float*)d_out;

  const int B = 4, Sc = 512, Se = 2048;
  const int Mc = B * Sc;   // 2048 claim tokens
  const int Me = B * Se;   // 8192 evidence tokens

  char* p = (char*)d_ws;
  auto carve = [&](size_t bytes) { char* q = p; p += bytes; return q; };
  __bf16* clb = (__bf16*)carve((size_t)Mc * HS * 2);
  __bf16* evb = (__bf16*)carve((size_t)Me * HS * 2);
  __bf16* wb[6];
  for (int i = 0; i < 6; ++i) wb[i] = (__bf16*)carve((size_t)HS * HS * 2);
  __bf16* Qc = (__bf16*)carve((size_t)Mc * HS * 2);
  __bf16* Ke = (__bf16*)carve((size_t)Me * HS * 2);
  __bf16* Ve = (__bf16*)carve((size_t)Me * HS * 2);
  __bf16* Qe = (__bf16*)carve((size_t)Me * HS * 2);
  __bf16* Kc = (__bf16*)carve((size_t)Mc * HS * 2);
  __bf16* Vc = (__bf16*)carve((size_t)Mc * HS * 2);
  float* attC = (float*)carve((size_t)Mc * HS * 4);
  float* attE = (float*)carve((size_t)Me * HS * 4);

  auto cvt = [&](const float* src, __bf16* dst, size_t n) {
    int blocks = (int)((n / 4 + 255) / 256);
    cvt_f32_bf16<<<blocks, 256, 0, stream>>>(src, dst, (int)n);
  };
  cvt(cl, clb, (size_t)Mc * HS);
  cvt(ev, evb, (size_t)Me * HS);
  const float* Wsrc[6] = {W_cq, W_ek, W_ev, W_eq, W_ck, W_cv};
  for (int i = 0; i < 6; ++i) cvt(Wsrc[i], wb[i], (size_t)HS * HS);

  dim3 blk(256);
  gemm_xwt_bf16<<<dim3(Mc / 64, HS / 128), blk, 0, stream>>>(clb, wb[0], b_cq, Qc, Mc);
  gemm_xwt_bf16<<<dim3(Me / 64, HS / 128), blk, 0, stream>>>(evb, wb[1], b_ek, Ke, Me);
  gemm_xwt_bf16<<<dim3(Me / 64, HS / 128), blk, 0, stream>>>(evb, wb[2], b_ev, Ve, Me);
  gemm_xwt_bf16<<<dim3(Me / 64, HS / 128), blk, 0, stream>>>(evb, wb[3], b_eq, Qe, Me);
  gemm_xwt_bf16<<<dim3(Mc / 64, HS / 128), blk, 0, stream>>>(clb, wb[4], b_ck, Kc, Mc);
  gemm_xwt_bf16<<<dim3(Mc / 64, HS / 128), blk, 0, stream>>>(clb, wb[5], b_cv, Vc, Mc);

  flash_attn_bf16<<<dim3(Sc / 64, NH, B), dim3(128), 0, stream>>>(Qc, Ke, Ve, emask, attC, Sc, Se);
  flash_attn_bf16<<<dim3(Se / 64, NH, B), dim3(128), 0, stream>>>(Qe, Kc, Vc, cmask, attE, Se, Sc);

  add_ln<<<Mc, 256, 0, stream>>>(cl, attC, ln_g, ln_b, out);
  add_ln<<<Me, 256, 0, stream>>>(ev, attE, ln_g, ln_b, out + (size_t)Mc * HS);
}
